// EagleAttention2_40596030882273
// MI455X (gfx1250) — matrix operand order repaired and checked by reading, not compiled, not run on hardware
//
#include <hip/hip_runtime.h>
#include <hip/hip_bf16.h>
#include <stdint.h>

typedef unsigned short ushort_t;
typedef __attribute__((ext_vector_type(16))) __bf16 v16bf;
typedef __attribute__((ext_vector_type(8)))  float  v8f;
typedef __attribute__((ext_vector_type(4)))  unsigned int u32x4;
typedef __attribute__((ext_vector_type(4)))  int v4i;

union Frag { v16bf v; u32x4 u[2]; };

__device__ __forceinline__ ushort_t f2bf(float f) {
    unsigned u = __builtin_bit_cast(unsigned, f);
    unsigned r = (u + 0x7FFFu + ((u >> 16) & 1u)) >> 16;
    return (ushort_t)r;
}

__device__ __forceinline__ v8f wmma_bf16(const Frag& a, const Frag& b, v8f c) {
    return __builtin_amdgcn_wmma_f32_16x16x32_bf16(false, a.v, false, b.v,
                                                   (short)0, c, false, false);
}

// ---- async global->LDS copy (CDNA5 path), with sync fallback -----------------
#if defined(__has_builtin)
#if __has_builtin(__builtin_amdgcn_global_load_async_to_lds_b128) && \
    __has_builtin(__builtin_amdgcn_s_wait_asynccnt)
#define HAVE_ASYNC_LDS 1
#endif
#endif

__device__ __forceinline__ void lds_copy16(void* l, const void* g) {
#ifdef HAVE_ASYNC_LDS
    __builtin_amdgcn_global_load_async_to_lds_b128(
        (v4i*)(uintptr_t)g, (v4i*)(uintptr_t)l, 0, 0);
#else
    *(u32x4*)l = *(const u32x4*)g;
#endif
}

__device__ __forceinline__ void lds_copy_wait() {
#ifdef HAVE_ASYNC_LDS
    __builtin_amdgcn_s_wait_asynccnt(0);
#endif
}

// ---------------------------------------------------------------- cast f32->bf16
__global__ void cast_f32_bf16(const float* __restrict__ in, ushort_t* __restrict__ out, int n) {
    int i = blockIdx.x * blockDim.x + threadIdx.x;
    if (i < n) out[i] = f2bf(in[i]);
}

// ---------------------------------------------------------------- generic GEMM: C = A @ W^T
// A: [M,K] bf16 row-major.  W: [N,K] bf16 row-major.
// mode 1: +RoPE per 64-wide head strip, store bf16 [M,N]
// mode 2: store bf16 transposed [N,M]
// mode 3: store f32 [M,N]
// else  : store bf16 [M,N]
// block = 128 threads (4 waves); block tile 16(M) x 256(N); A tile staged in LDS
// (double-buffered async copy shared by the 4 waves), B streamed from global.
__global__ __launch_bounds__(128) void gemm_bf16(
    const ushort_t* __restrict__ A, const ushort_t* __restrict__ W,
    int M, int N, int K, int mode,
    ushort_t* __restrict__ outB, float* __restrict__ outF)
{
    __shared__ __align__(16) ushort_t aTile[2][16 * 64];   // 2 stages x 2KB

    const int tid  = threadIdx.x;
    const int lane = tid & 31;
    const int wave = tid >> 5;
    const int l15  = lane & 15;
    const int hi   = lane >> 4;
    const int mbase = blockIdx.y * 16;
    const int nbase = blockIdx.x * 256 + wave * 64;

    // A tile staging: thread t copies one 16B chunk: row = t/8, 8 elems at (t%8)*8
    const int arow = tid >> 3;
    const int aoff = (tid & 7) * 8;
    const ushort_t* aSrc = A + (size_t)(mbase + arow) * K + aoff;

    auto stageA = [&](int k0, int stg) {
        lds_copy16(&aTile[stg][arow * 64 + aoff], aSrc + k0);
    };

    v8f acc[4] = {};
    stageA(0, 0);

    for (int k0 = 0; k0 < K; k0 += 64) {
        const int stg = (k0 >> 6) & 1;
        lds_copy_wait();
        __syncthreads();
        if (k0 + 64 < K) stageA(k0 + 64, stg ^ 1);     // prefetch next stage

#pragma unroll
        for (int c = 0; c < 2; ++c) {                  // two K=32 WMMA steps
            Frag af;
            const ushort_t* ap = &aTile[stg][l15 * 64 + c * 32 + hi * 8];
            af.u[0] = *(const u32x4*)(ap);
            af.u[1] = *(const u32x4*)(ap + 16);
#pragma unroll
            for (int j = 0; j < 4; ++j) {
                const ushort_t* wrow = W + (size_t)(nbase + j * 16 + l15) * K
                                         + (k0 + c * 32) + hi * 8;
                Frag bf;
                bf.u[0] = *(const u32x4*)(wrow);
                bf.u[1] = *(const u32x4*)(wrow + 16);
                acc[j] = wmma_bf16(af, bf, acc[j]);
            }
        }
    }

    if (mode == 1) {
        // RoPE: strip is exactly one head (64 wide). Pair (d, d+32) = (acc[j], acc[j+2]).
        const float NEG_LOG_BASE_OVER_32 = -0.28782313662425572f; // -ln(10000)/32
#pragma unroll
        for (int j = 0; j < 2; ++j) {
            int i = j * 16 + l15;                       // freq index = d % 32
            float inv = __expf((float)i * NEG_LOG_BASE_OVER_32);
#pragma unroll
            for (int r = 0; r < 8; ++r) {
                int m = mbase + r + 8 * hi;             // position id
                float ang = (float)m * inv;
                float c = __cosf(ang), s = __sinf(ang);
                float x1 = acc[j][r], x2 = acc[j + 2][r];
                acc[j][r]     = x1 * c - x2 * s;
                acc[j + 2][r] = x2 * c + x1 * s;
            }
        }
    }

    if (nbase < N) {
#pragma unroll
        for (int j = 0; j < 4; ++j) {
#pragma unroll
            for (int r = 0; r < 8; ++r) {
                int m = mbase + r + 8 * hi;
                int n = nbase + j * 16 + l15;
                if (mode == 2)      outB[(size_t)n * M + m] = f2bf(acc[j][r]);
                else if (mode == 3) outF[(size_t)m * N + n] = acc[j][r];
                else                outB[(size_t)m * N + n] = f2bf(acc[j][r]);
            }
        }
    }
}

// ---------------------------------------------------------------- flash attention
// Block = (head, 64-row band), 4 waves x 16 rows. K (32x64) and V (64x32) tiles
// double-buffered in LDS via async copies, shared by all waves.
// Q:[2048,1024] bf16. Kc:[2048,256] bf16. Vt:[256,2048] bf16. Oout:[2048,1024] bf16.
__global__ __launch_bounds__(128) void attn_kernel(
    const ushort_t* __restrict__ Q, const ushort_t* __restrict__ Kc,
    const ushort_t* __restrict__ Vt, ushort_t* __restrict__ Oout)
{
    const int tid  = threadIdx.x;
    const int lane = tid & 31;
    const int wave = tid >> 5;
    const int l15  = lane & 15;
    const int hi   = lane >> 4;
    const int h     = blockIdx.x;           // 0..15
    const int mband = blockIdx.y * 64;      // 64-row band
    const int mbase = mband + wave * 16;    // this wave's 16 rows
    const int hkv   = h >> 2;               // GQA group

    __shared__ __align__(16) ushort_t kbuf[2][32 * 64];  // keys x d
    __shared__ __align__(16) ushort_t vbuf[2][64 * 32];  // d x keys (from Vt)
    __shared__ __align__(16) ushort_t pshm[4][16 * 32];  // per-wave P transpose

    // stage one 32-key tile of K and V (4KB each): 256 16B-chunks apiece,
    // each of the 128 threads moves 2 chunks of each.
    auto stageKV = [&](int kb, int stg) {
#pragma unroll
        for (int c = tid; c < 256; c += 128) {
            const int krow = c >> 3, koff = (c & 7) * 8;   // key row, d offset
            lds_copy16(&kbuf[stg][krow * 64 + koff],
                       Kc + (size_t)(kb + krow) * 256 + hkv * 64 + koff);
            const int vrow = c >> 2, voff = (c & 3) * 8;   // d row, key offset
            lds_copy16(&vbuf[stg][vrow * 32 + voff],
                       Vt + (size_t)(hkv * 64 + vrow) * 2048 + kb + voff);
        }
    };

    // Q fragments for d in [0,64): two K=32 A-fragments (per wave, from global)
    const ushort_t* qrow = Q + (size_t)(mbase + l15) * 1024 + h * 64 + hi * 8;
    Frag qa[2];
    qa[0].u[0] = *(const u32x4*)(qrow);
    qa[0].u[1] = *(const u32x4*)(qrow + 16);
    qa[1].u[0] = *(const u32x4*)(qrow + 32);
    qa[1].u[1] = *(const u32x4*)(qrow + 48);

    v8f o[4] = {};
    float mrow[8], lrow[8];
#pragma unroll
    for (int r = 0; r < 8; ++r) { mrow[r] = -3.0e38f; lrow[r] = 0.0f; }

    const int nkt = mband / 32 + 2;          // keys [0, mband+64) in 32-key tiles
    stageKV(0, 0);

    for (int kt = 0; kt < nkt; ++kt) {
        const int kb  = kt * 32;
        const int stg = kt & 1;
        lds_copy_wait();
        __syncthreads();
        if (kt + 1 < nkt) stageKV(kb + 32, stg ^ 1);   // prefetch next tile

        if (kb < mbase + 16) {                         // wave's causal limit
            // S = Q K^T from LDS K tile
            v8f s[2] = {};
#pragma unroll
            for (int j = 0; j < 2; ++j) {
#pragma unroll
                for (int c = 0; c < 2; ++c) {
                    const ushort_t* kp = &kbuf[stg][(j * 16 + l15) * 64 + c * 32 + hi * 8];
                    Frag bk;
                    bk.u[0] = *(const u32x4*)(kp);
                    bk.u[1] = *(const u32x4*)(kp + 16);
                    s[j] = wmma_bf16(qa[c], bk, s[j]);
                }
            }

            // scale + causal mask + online softmax
#pragma unroll
            for (int r = 0; r < 8; ++r) {
                int row = mbase + r + 8 * hi;
#pragma unroll
                for (int j = 0; j < 2; ++j) {
                    int col = kb + j * 16 + l15;
                    float v = s[j][r] * 0.125f;        // 1/sqrt(64)
                    if (col > row) v += -1.0e9f;       // causal (matches reference)
                    s[j][r] = v;
                }
                float mx = fmaxf(s[0][r], s[1][r]);
#pragma unroll
                for (int off = 1; off < 16; off <<= 1)
                    mx = fmaxf(mx, __shfl_xor(mx, off, 32));   // within half-wave
                float mnew = fmaxf(mrow[r], mx);
                float alpha = __expf(mrow[r] - mnew);
                float psum = 0.0f;
#pragma unroll
                for (int j = 0; j < 2; ++j) {
                    float p = __expf(s[j][r] - mnew);
                    s[j][r] = p;
                    psum += p;
                }
#pragma unroll
                for (int off = 1; off < 16; off <<= 1)
                    psum += __shfl_xor(psum, off, 32);
                lrow[r] = lrow[r] * alpha + psum;
                mrow[r] = mnew;
#pragma unroll
                for (int jj = 0; jj < 4; ++jj) o[jj][r] *= alpha;
            }

            // P: C-layout -> A-layout through wave-private LDS (same-wave DS is in-order)
#pragma unroll
            for (int r = 0; r < 8; ++r) {
#pragma unroll
                for (int j = 0; j < 2; ++j)
                    pshm[wave][(r + 8 * hi) * 32 + j * 16 + l15] = f2bf(s[j][r]);
            }
            Frag pa;
            const ushort_t* prow = &pshm[wave][l15 * 32 + hi * 8];
            pa.u[0] = *(const u32x4*)(prow);
            pa.u[1] = *(const u32x4*)(prow + 16);

            // O += P @ V from LDS V tile
#pragma unroll
            for (int j2 = 0; j2 < 4; ++j2) {
                const ushort_t* vp = &vbuf[stg][(j2 * 16 + l15) * 32 + hi * 8];
                Frag bv;
                bv.u[0] = *(const u32x4*)(vp);
                bv.u[1] = *(const u32x4*)(vp + 16);
                o[j2] = wmma_bf16(pa, bv, o[j2]);
            }
        }
    }

    // normalize and store bf16
#pragma unroll
    for (int j2 = 0; j2 < 4; ++j2) {
#pragma unroll
        for (int r = 0; r < 8; ++r) {
            int row = mbase + r + 8 * hi;
            int col = h * 64 + j2 * 16 + l15;
            Oout[(size_t)row * 1024 + col] = f2bf(o[j2][r] / lrow[r]);
        }
    }
}

// ---------------------------------------------------------------- launcher
extern "C" void kernel_launch(void* const* d_in, const int* in_sizes, int n_in,
                              void* d_out, int out_size, void* d_ws, size_t ws_size,
                              hipStream_t stream) {
    (void)in_sizes; (void)n_in; (void)out_size; (void)ws_size;
    const int S = 2048, HID = 1024, NKVD = 256;

    const float* hidden = (const float*)d_in[0];
    // d_in[1] = attention_mask (causal, implemented analytically)
    // d_in[2] = position_ids   (arange, implemented analytically)
    const float* Wq = (const float*)d_in[3];
    const float* Wk = (const float*)d_in[4];
    const float* Wv = (const float*)d_in[5];
    const float* Wo = (const float*)d_in[6];

    auto alignup = [](size_t x) { return (x + 255) & ~(size_t)255; };
    char* p = (char*)d_ws;
    ushort_t* Xb  = (ushort_t*)p; p += alignup((size_t)S * HID * 2);
    ushort_t* Wqb = (ushort_t*)p; p += alignup((size_t)HID * HID * 2);
    ushort_t* Wkb = (ushort_t*)p; p += alignup((size_t)NKVD * HID * 2);
    ushort_t* Wvb = (ushort_t*)p; p += alignup((size_t)NKVD * HID * 2);
    ushort_t* Wob = (ushort_t*)p; p += alignup((size_t)HID * HID * 2);
    ushort_t* Qr  = (ushort_t*)p; p += alignup((size_t)S * HID * 2);
    ushort_t* Kr  = (ushort_t*)p; p += alignup((size_t)S * NKVD * 2);
    ushort_t* Vt  = (ushort_t*)p; p += alignup((size_t)NKVD * S * 2);
    ushort_t* AO  = (ushort_t*)p; p += alignup((size_t)S * HID * 2);

    auto cast = [&](const float* src, ushort_t* dst, int n) {
        cast_f32_bf16<<<(n + 255) / 256, 256, 0, stream>>>(src, dst, n);
    };
    cast(hidden, Xb,  S * HID);
    cast(Wq,     Wqb, HID * HID);
    cast(Wk,     Wkb, NKVD * HID);
    cast(Wv,     Wvb, NKVD * HID);
    cast(Wo,     Wob, HID * HID);

    // Q projection + RoPE: [2048,1024]
    gemm_bf16<<<dim3(HID / 256, S / 16), 128, 0, stream>>>(
        Xb, Wqb, S, HID, HID, 1, Qr, nullptr);
    // K projection + RoPE: [2048,256]
    gemm_bf16<<<dim3(1, S / 16), 128, 0, stream>>>(
        Xb, Wkb, S, NKVD, HID, 1, Kr, nullptr);
    // V projection, transposed store: Vt [256,2048]
    gemm_bf16<<<dim3(1, S / 16), 128, 0, stream>>>(
        Xb, Wvb, S, NKVD, HID, 2, Vt, nullptr);

    // flash attention: 16 heads x 32 row-bands, 4 waves/block
    attn_kernel<<<dim3(16, S / 64), 128, 0, stream>>>(Qr, Kr, Vt, AO);

    // output projection (f32 out): [2048,1024]
    gemm_bf16<<<dim3(HID / 256, S / 16), 128, 0, stream>>>(
        AO, Wob, S, HID, HID, 3, nullptr, (float*)d_out);
}
